// ReferenceFineGrainedEncoder_32882269618232
// MI455X (gfx1250) — compile-verified, matrix operand order checked by reading, not run on hardware
//
#include <hip/hip_runtime.h>
#include <hip/hip_bf16.h>

// ---------------- problem constants ----------------
constexpr int NB   = 16;    // batch
constexpr int TT   = 512;   // time
constexpr int FDIM = 80;    // feature dim
constexpr int PP   = 128;   // phones
constexpr int GRUH = 32;
constexpr int DMAX = 4;
constexpr int CFIN = 64;                 // final conv channels
constexpr int IN_F = CFIN * FDIM;        // 5120
constexpr int SEQ  = NB * PP;            // 2048
constexpr int ROWS = SEQ * DMAX;         // 8192
constexpr int G3   = 3 * GRUH;           // 96
constexpr float EPS = 1e-5f;

typedef __attribute__((ext_vector_type(16))) __bf16 bf16x16;
typedef __attribute__((ext_vector_type(8)))  float  f32x8;

__device__ __forceinline__ unsigned short f32_to_bf16_rne(float f) {
    unsigned int u = __float_as_uint(f);
    unsigned int r = u + 0x7FFFu + ((u >> 16) & 1u);
    return (unsigned short)(r >> 16);
}

// ---------------- conv 3x3 SAME + bias ----------------
template<int CIN, int COUT>
__global__ void conv3x3_bias(const float* __restrict__ x, const float* __restrict__ w,
                             const float* __restrict__ b, float* __restrict__ y) {
    __shared__ float wsm[CIN * 9];
    const int t  = blockIdx.x;
    const int co = blockIdx.y;
    const int n  = blockIdx.z;
    for (int i = threadIdx.x; i < CIN * 9; i += blockDim.x)
        wsm[i] = w[co * CIN * 9 + i];
    __syncthreads();
    const int f = threadIdx.x;
    if (f >= FDIM) return;
    float acc = b[co];
    for (int ci = 0; ci < CIN; ++ci) {
        const float* xp = x + ((size_t)(n * CIN + ci) * TT) * FDIM;
        const float* wp = wsm + ci * 9;
#pragma unroll
        for (int dt = -1; dt <= 1; ++dt) {
            int tt = t + dt;
            if (tt < 0 || tt >= TT) continue;
#pragma unroll
            for (int df = -1; df <= 1; ++df) {
                int ff = f + df;
                if (ff < 0 || ff >= FDIM) continue;
                acc += xp[tt * FDIM + ff] * wp[(dt + 1) * 3 + (df + 1)];
            }
        }
    }
    y[((size_t)(n * COUT + co) * TT + t) * FDIM + f] = acc;
}

// ---------------- batchnorm statistics (one block per channel) ----------------
__global__ void bn_stats(const float* __restrict__ x, int C, float* __restrict__ stats) {
    const int c = blockIdx.x;
    const int per = TT * FDIM;
    float s = 0.f, sq = 0.f;
    for (int i = threadIdx.x; i < NB * per; i += blockDim.x) {
        int n = i / per;
        int r = i % per;
        float v = x[((size_t)(n * C + c)) * per + r];
        s += v; sq += v * v;
    }
    __shared__ float ssum[256], ssq[256];
    ssum[threadIdx.x] = s; ssq[threadIdx.x] = sq;
    __syncthreads();
    for (int o = 128; o > 0; o >>= 1) {
        if ((int)threadIdx.x < o) {
            ssum[threadIdx.x] += ssum[threadIdx.x + o];
            ssq[threadIdx.x]  += ssq[threadIdx.x + o];
        }
        __syncthreads();
    }
    if (threadIdx.x == 0) {
        float m = (float)NB * per;
        float mean = ssum[0] / m;
        float var  = ssq[0] / m - mean * mean;
        stats[c * 2]     = mean;
        stats[c * 2 + 1] = var;
    }
}

// ---------------- normalize + affine + relu (in place) ----------------
__global__ void bn_apply_relu(float* __restrict__ x, int C, const float* __restrict__ stats,
                              const float* __restrict__ g, const float* __restrict__ bt) {
    const long total = (long)NB * C * TT * FDIM;
    const int per = TT * FDIM;
    for (long i = blockIdx.x * (long)blockDim.x + threadIdx.x; i < total;
         i += (long)gridDim.x * blockDim.x) {
        int c = (int)((i / per) % C);
        float v = (x[i] - stats[c * 2]) * rsqrtf(stats[c * 2 + 1] + EPS);
        v = v * g[c] + bt[c];
        x[i] = fmaxf(v, 0.f);
    }
}

// ---------------- f32 -> bf16 conversion ----------------
__global__ void cvt_bf16(const float* __restrict__ src, unsigned short* __restrict__ dst, int n) {
    int i = blockIdx.x * blockDim.x + threadIdx.x;
    if (i < n) dst[i] = f32_to_bf16_rne(src[i]);
}

// ---------------- segment gather -> xseq / rxseq (bf16) ----------------
__global__ void gather_segments(const float* __restrict__ x4,   // [NB][64][TT][FDIM]
                                const int* __restrict__ ds,     // [NB*PP]
                                unsigned short* __restrict__ xseq,
                                unsigned short* __restrict__ rxseq) {
    const int s = blockIdx.x;
    const int n = s / PP, p = s % PP;
    int start = 0;
    for (int j = 0; j < p; ++j) start += ds[n * PP + j];
    const int d = ds[n * PP + p];
    const int L = d > 0 ? d : 1;
#pragma unroll
    for (int t = 0; t < DMAX; ++t) {
        // forward row t
        {
            int row = start + t;
            row = row < 0 ? 0 : (row > TT - 1 ? TT - 1 : row);
            const bool valid = t < d;
            unsigned short* dst = xseq + ((size_t)(s * DMAX + t)) * IN_F;
            for (int i = threadIdx.x; i < IN_F; i += blockDim.x) {
                int c = i / FDIM, f = i % FDIM;
                float v = valid ? x4[((size_t)(n * CFIN + c) * TT + row) * FDIM + f] : 0.f;
                dst[i] = f32_to_bf16_rne(v);
            }
        }
        // reverse row t = xseq row clip(L-1-t, 0, DMAX-1), computed directly
        {
            int u = L - 1 - t;
            u = u < 0 ? 0 : (u > DMAX - 1 ? DMAX - 1 : u);
            int row = start + u;
            row = row < 0 ? 0 : (row > TT - 1 ? TT - 1 : row);
            const bool valid = u < d;
            unsigned short* dst = rxseq + ((size_t)(s * DMAX + t)) * IN_F;
            for (int i = threadIdx.x; i < IN_F; i += blockDim.x) {
                int c = i / FDIM, f = i % FDIM;
                float v = valid ? x4[((size_t)(n * CFIN + c) * TT + row) * FDIM + f] : 0.f;
                dst[i] = f32_to_bf16_rne(v);
            }
        }
    }
}

// ---------------- WMMA GEMM: C[ROWS][96] = A[ROWS][5120] * W[96][5120]^T + bias ----------------
// 128 threads = 4 waves; each wave owns TWO 16-row M tiles and all 6 N tiles.
// The 96x32 bf16 B-panel for the current K step is staged in LDS once per block
// and shared by all waves; each B fragment feeds two back-to-back WMMAs.
// __launch_bounds__(128, 1): allow a full VGPR budget so the 12 f32x8
// accumulators stay in registers (no inner-loop scratch spills).
__global__ void __launch_bounds__(128, 1)
gemm_wmma(const unsigned short* __restrict__ Abf,   // [ROWS][IN_F] bf16
          const unsigned short* __restrict__ Wbf,   // [G3][IN_F]  bf16
          const float* __restrict__ bias,           // [G3]
          float* __restrict__ C) {                  // [ROWS][G3]
    __shared__ __align__(16) __bf16 Bs[G3 * 32];   // 6 KB: W[:, k0:k0+32]
    const int wave  = threadIdx.x >> 5;
    const int lane  = threadIdx.x & 31;
    const int lcol  = lane & 15;
    const int khalf = (lane >> 4) ? 8 : 0;          // high half-wave holds K+8 / K+24 groups
    const int mbase = blockIdx.x * 128 + wave * 32; // two M tiles: mbase, mbase+16

    f32x8 acc[2][6] = {};
    const __bf16* arow0 = (const __bf16*)Abf + (size_t)(mbase + lcol) * IN_F;
    const __bf16* arow1 = arow0 + (size_t)16 * IN_F;

    for (int k0 = 0; k0 < IN_F; k0 += 32) {
        __syncthreads();   // protect previous iteration's Bs reads
        // stage W[:, k0:k0+32] into LDS with 16-byte vector copies (96 rows x 4 chunks)
        for (int chunk = threadIdx.x; chunk < G3 * 4; chunk += blockDim.x) {
            const int c = chunk >> 2, part = chunk & 3;
            const uint4 v = *(const uint4*)((const __bf16*)Wbf + (size_t)c * IN_F + k0 + part * 8);
            *(uint4*)(&Bs[c * 32 + part * 8]) = v;
        }
        // prefetch next K step of A while the LDS fill is in flight
        if (k0 + 32 < IN_F) {
            __builtin_prefetch(arow0 + k0 + 32, 0, 0);
            __builtin_prefetch(arow1 + k0 + 32, 0, 0);
        }
        // A fragments (2 x b128 global loads each)
        bf16x16 a0, a1;
#pragma unroll
        for (int i = 0; i < 8; ++i) {
            a0[i]     = arow0[k0 + khalf + i];      // VGPR0..3: K 0..7 (lo lanes) / 8..15 (hi)
            a0[8 + i] = arow0[k0 + khalf + 16 + i]; // VGPR4..7: K 16..23 / 24..31
            a1[i]     = arow1[k0 + khalf + i];
            a1[8 + i] = arow1[k0 + khalf + 16 + i];
        }
        __syncthreads();   // Bs ready
#pragma unroll
        for (int j = 0; j < 6; ++j) {
            const __bf16* bp = &Bs[(j * 16 + lcol) * 32];
            bf16x16 bfrag;
#pragma unroll
            for (int i = 0; i < 8; ++i) {
                bfrag[i]     = bp[khalf + i];
                bfrag[8 + i] = bp[khalf + 16 + i];
            }
            acc[0][j] = __builtin_amdgcn_wmma_f32_16x16x32_bf16(
                false, a0, false, bfrag, (short)0, acc[0][j], false, false);
            acc[1][j] = __builtin_amdgcn_wmma_f32_16x16x32_bf16(
                false, a1, false, bfrag, (short)0, acc[1][j], false, false);
        }
    }
    // epilogue: C/D layout: VGPR r -> M = r (lanes 0-15) / M = 8+r (lanes 16-31); N = lane%16
    const int mtop = (lane >> 4) ? 8 : 0;
#pragma unroll
    for (int u = 0; u < 2; ++u) {
#pragma unroll
        for (int j = 0; j < 6; ++j) {
            const int col = j * 16 + lcol;
            const float bv = bias[col];
#pragma unroll
            for (int r = 0; r < 8; ++r) {
                const int m = mbase + u * 16 + mtop + r;
                C[(size_t)m * G3 + col] = acc[u][j][r] + bv;
            }
        }
    }
}

// ---------------- bidirectional GRU, one wave32 per sequence ----------------
__device__ __forceinline__ float sigmoidf_(float x) { return 1.f / (1.f + __expf(-x)); }

__global__ void gru_bidir(const float* __restrict__ gi_f, const float* __restrict__ gi_b,
                          const float* __restrict__ whh_f, const float* __restrict__ bhh_f,
                          const float* __restrict__ whh_b, const float* __restrict__ bhh_b,
                          const int* __restrict__ ds, const int* __restrict__ lens,
                          float* __restrict__ outp) {
    __shared__ float wf[G3 * GRUH];
    __shared__ float wb[G3 * GRUH];
    const int s = blockIdx.x;
    const int lane = threadIdx.x;           // blockDim == 32
    for (int i = lane; i < G3 * GRUH; i += 32) { wf[i] = whh_f[i]; wb[i] = whh_b[i]; }
    __syncthreads();

    const int n = s / PP, p = s % PP;
    const int d = ds[s];
    const int L = d > 0 ? d : 1;
    const bool pvalid = p < lens[n];

    float h = 0.f;
#pragma unroll
    for (int t = 0; t < DMAX; ++t) {
        const float* gi = gi_f + ((size_t)(s * DMAX + t)) * G3;
        float gr = bhh_f[lane], gz = bhh_f[GRUH + lane], gn = bhh_f[2 * GRUH + lane];
        for (int k = 0; k < GRUH; ++k) {
            float hv = __shfl(h, k, 32);
            gr += hv * wf[lane * GRUH + k];
            gz += hv * wf[(GRUH + lane) * GRUH + k];
            gn += hv * wf[(2 * GRUH + lane) * GRUH + k];
        }
        float r  = sigmoidf_(gi[lane] + gr);
        float z  = sigmoidf_(gi[GRUH + lane] + gz);
        float nn = tanhf(gi[2 * GRUH + lane] + r * gn);
        float hnew = (1.f - z) * nn + z * h;
        if (t < L) h = hnew;
    }
    const float hf = h;

    h = 0.f;
#pragma unroll
    for (int t = 0; t < DMAX; ++t) {
        const float* gi = gi_b + ((size_t)(s * DMAX + t)) * G3;
        float gr = bhh_b[lane], gz = bhh_b[GRUH + lane], gn = bhh_b[2 * GRUH + lane];
        for (int k = 0; k < GRUH; ++k) {
            float hv = __shfl(h, k, 32);
            gr += hv * wb[lane * GRUH + k];
            gz += hv * wb[(GRUH + lane) * GRUH + k];
            gn += hv * wb[(2 * GRUH + lane) * GRUH + k];
        }
        float r  = sigmoidf_(gi[lane] + gr);
        float z  = sigmoidf_(gi[GRUH + lane] + gz);
        float nn = tanhf(gi[2 * GRUH + lane] + r * gn);
        float hnew = (1.f - z) * nn + z * h;
        if (t < L) h = hnew;
    }
    const float hb = h;

    float* o = outp + ((size_t)(n * PP + p)) * (2 * GRUH);
    o[lane]        = pvalid ? hf : 0.f;
    o[GRUH + lane] = pvalid ? hb : 0.f;
}

// ---------------- orchestration ----------------
extern "C" void kernel_launch(void* const* d_in, const int* in_sizes, int n_in,
                              void* d_out, int out_size, void* d_ws, size_t ws_size,
                              hipStream_t stream) {
    const float* inputs = (const float*)d_in[0];
    const int*   ds     = (const int*)d_in[1];
    const int*   lens   = (const int*)d_in[2];
    const float* cw[4]  = {(const float*)d_in[3], (const float*)d_in[4], (const float*)d_in[5], (const float*)d_in[6]};
    const float* cb[4]  = {(const float*)d_in[7], (const float*)d_in[8], (const float*)d_in[9], (const float*)d_in[10]};
    const float* bg[4]  = {(const float*)d_in[11], (const float*)d_in[12], (const float*)d_in[13], (const float*)d_in[14]};
    const float* bb[4]  = {(const float*)d_in[15], (const float*)d_in[16], (const float*)d_in[17], (const float*)d_in[18]};
    const float* w_ih_f = (const float*)d_in[19];
    const float* w_hh_f = (const float*)d_in[20];
    const float* b_ih_f = (const float*)d_in[21];
    const float* b_hh_f = (const float*)d_in[22];
    const float* w_ih_b = (const float*)d_in[23];
    const float* w_hh_b = (const float*)d_in[24];
    const float* b_ih_b = (const float*)d_in[25];
    const float* b_hh_b = (const float*)d_in[26];

    // workspace layout (bytes)
    char* w8 = (char*)d_ws;
    const size_t BUF = (size_t)NB * CFIN * TT * FDIM * sizeof(float);   // 167,772,160
    float* bufA  = (float*)w8;
    float* bufB  = (float*)(w8 + BUF);
    float* stats = (float*)(w8 + 2 * BUF);                              // 128 f32
    unsigned short* WbF  = (unsigned short*)(w8 + 2 * BUF + 1024);
    unsigned short* WbB  = (unsigned short*)((char*)WbF + (size_t)G3 * IN_F * 2);
    float* giF = (float*)((char*)WbB + (size_t)G3 * IN_F * 2);
    float* giB = (float*)((char*)giF + (size_t)ROWS * G3 * 4);
    // xseq/rxseq reuse bufA region once the conv stack has finished (final data in bufB)
    unsigned short* xseq  = (unsigned short*)bufA;
    unsigned short* rxseq = (unsigned short*)((char*)bufA + (size_t)ROWS * IN_F * 2);

    const int ELEM_BLOCKS = 4096;

    // layer 1: inputs (cin=1) -> bufA
    conv3x3_bias<1, 32><<<dim3(TT, 32, NB), 128, 0, stream>>>(inputs, cw[0], cb[0], bufA);
    bn_stats<<<32, 256, 0, stream>>>(bufA, 32, stats);
    bn_apply_relu<<<ELEM_BLOCKS, 256, 0, stream>>>(bufA, 32, stats, bg[0], bb[0]);
    // layer 2: bufA -> bufB
    conv3x3_bias<32, 32><<<dim3(TT, 32, NB), 128, 0, stream>>>(bufA, cw[1], cb[1], bufB);
    bn_stats<<<32, 256, 0, stream>>>(bufB, 32, stats);
    bn_apply_relu<<<ELEM_BLOCKS, 256, 0, stream>>>(bufB, 32, stats, bg[1], bb[1]);
    // layer 3: bufB -> bufA
    conv3x3_bias<32, 64><<<dim3(TT, 64, NB), 128, 0, stream>>>(bufB, cw[2], cb[2], bufA);
    bn_stats<<<64, 256, 0, stream>>>(bufA, 64, stats);
    bn_apply_relu<<<ELEM_BLOCKS, 256, 0, stream>>>(bufA, 64, stats, bg[2], bb[2]);
    // layer 4: bufA -> bufB (final activations live in bufB)
    conv3x3_bias<64, 64><<<dim3(TT, 64, NB), 128, 0, stream>>>(bufA, cw[3], cb[3], bufB);
    bn_stats<<<64, 256, 0, stream>>>(bufB, 64, stats);
    bn_apply_relu<<<ELEM_BLOCKS, 256, 0, stream>>>(bufB, 64, stats, bg[3], bb[3]);

    // weights to bf16
    {
        int nW = G3 * IN_F;
        int blocks = (nW + 255) / 256;
        cvt_bf16<<<blocks, 256, 0, stream>>>(w_ih_f, WbF, nW);
        cvt_bf16<<<blocks, 256, 0, stream>>>(w_ih_b, WbB, nW);
    }

    // gather segments into bf16 xseq / rxseq (reads bufB, writes bufA region)
    gather_segments<<<SEQ, 256, 0, stream>>>(bufB, ds, xseq, rxseq);

    // WMMA GEMMs: gi = xseq @ W^T + b_ih   (128 rows per block)
    gemm_wmma<<<ROWS / 128, 128, 0, stream>>>(xseq,  WbF, b_ih_f, giF);
    gemm_wmma<<<ROWS / 128, 128, 0, stream>>>(rxseq, WbB, b_ih_b, giB);

    // bidirectional GRU + final masking -> d_out (N, P, 64)
    gru_bidir<<<SEQ, 32, 0, stream>>>(giF, giB, w_hh_f, b_hh_f, w_hh_b, b_hh_b,
                                      ds, lens, (float*)d_out);
}
